// EfficientAttention_31533649887315
// MI455X (gfx1250) — matrix operand order repaired and checked by reading, not compiled, not run on hardware
//
#include <hip/hip_runtime.h>

typedef __attribute__((ext_vector_type(16))) _Float16 v16h;
typedef __attribute__((ext_vector_type(8)))  _Float16 v8h;
typedef __attribute__((ext_vector_type(8)))  float    v8f;

union AFrag { v16h v; struct { v8h lo; v8h hi; } h; };

#define WMMA_F16(A, B, C) \
  __builtin_amdgcn_wmma_f32_16x16x32_f16(false, (A), false, (B), (short)0, (C), false, false)

// Async copy of one 16-byte chunk: global -> LDS (ASYNCcnt-tracked, gfx1250)
__device__ __forceinline__ void async_g2l_b128(const _Float16* gsrc, void* ldst) {
  unsigned ldsoff = (unsigned)(unsigned long long)ldst;  // low 32b of LDS aperture addr
  asm volatile("global_load_async_to_lds_b128 %0, %1, off"
               :: "v"(ldsoff), "v"(gsrc) : "memory");
}

// --------------------------------------------------------------------------
// f32 -> f16 conversion
// --------------------------------------------------------------------------
__global__ void cvt_f16_kernel(const float* __restrict__ src,
                               _Float16* __restrict__ dst, int n) {
  int i = blockIdx.x * blockDim.x + threadIdx.x;
  int stride = gridDim.x * blockDim.x;
  for (; i < n; i += stride) dst[i] = (_Float16)src[i];
}

// --------------------------------------------------------------------------
// QKV projection + fused RoPE (q,k) / fused transpose store (v)
// grid: (32 row tiles of 128, 32 col pairs of 16, 3 matrices), block 256 (8 waves)
// Weight tile (32 rows x 32 k) staged in LDS via async-to-LDS, double buffered.
// --------------------------------------------------------------------------
__global__ __launch_bounds__(256)
void qkv_kernel(const _Float16* __restrict__ xh,
                const _Float16* __restrict__ wqh,
                const _Float16* __restrict__ wkh,
                const _Float16* __restrict__ wvh,
                _Float16* __restrict__ Qr,
                _Float16* __restrict__ Kr,
                _Float16* __restrict__ Vt) {
  __shared__ __align__(128) _Float16 bstage[2][32][32];  // 2 x 2KB weight tiles
  const int tid  = threadIdx.x;
  const int lane = tid & 31;
  const int wave = tid >> 5;
  const int lr = lane & 15, lh = lane >> 4;
  const int rowbase = blockIdx.x * 128 + wave * 16;
  const int n0 = blockIdx.y * 16;
  const int mtx = blockIdx.z;
  const _Float16* wh = (mtx == 0) ? wqh : (mtx == 1) ? wkh : wvh;

  const _Float16* aptr = xh + (size_t)(rowbase + lr) * 1024 + lh * 16;

  // Async-copy lane assignment: threads 0..127 move 128 x 16B = one 32x32 tile.
  const bool cp  = tid < 128;
  const int crow = tid >> 2;          // 0..31 (staged row)
  const int cc   = (tid & 3) * 8;     // element offset within row (0,8,16,24)
  const int grow = (crow < 16) ? (n0 + crow) : (n0 + 496 + crow);  // cols d / d+512
  const _Float16* gsrc = wh + (size_t)grow * 1024 + cc;

  if (cp) async_g2l_b128(gsrc, &bstage[0][crow][cc]);   // prologue: batch 0

  v8f acc0 = {}, acc1 = {};
  for (int kk = 0; kk < 1024; kk += 32) {
    const int cur = (kk >> 5) & 1;
    if (cp) {
      if (kk + 32 < 1024) {
        async_g2l_b128(gsrc + kk + 32, &bstage[cur ^ 1][crow][cc]);
        asm volatile("s_wait_asynccnt 0x1" ::: "memory");  // batch kk has landed
      } else {
        asm volatile("s_wait_asynccnt 0x0" ::: "memory");
      }
    }
    __syncthreads();

    __builtin_prefetch(aptr + kk + 256, 0, 3);  // global_prefetch_b8 (near)
    AFrag a;
    a.h.lo = *(const v8h*)(aptr + kk);
    a.h.hi = *(const v8h*)(aptr + kk + 8);
    v16h b0 = *(const v16h*)&bstage[cur][lr][lh * 16];
    v16h b1 = *(const v16h*)&bstage[cur][16 + lr][lh * 16];
    acc0 = WMMA_F16(a.v, b0, acc0);
    acc1 = WMMA_F16(a.v, b1, acc1);
    __syncthreads();  // protect bstage[cur] until all waves consumed it
  }

  const int row0 = rowbase + 8 * lh;          // first of 8 rows this lane holds
  if (mtx < 2) {
    // Fused RoPE: pair (d, d+512), ang = pos * 10000^(-d/512)
    _Float16* dst = (mtx == 0) ? Qr : Kr;
    const int d = n0 + lr;
    const float freq = __expf(-(float)d * (9.210340371976184f / 512.0f));
#pragma unroll
    for (int r = 0; r < 8; ++r) {
      const int srow = row0 + r;               // global row in [0,4096)
      const float pos = (float)(srow & 2047);  // sequence position
      const float ang = pos * freq;
      const float sn = __sinf(ang), cs = __cosf(ang);
      const float x1 = acc0[r], x2 = acc1[r];
      dst[(size_t)srow * 1024 + d]       = (_Float16)(x1 * cs - x2 * sn);
      dst[(size_t)srow * 1024 + d + 512] = (_Float16)(x1 * sn + x2 * cs);
    }
  } else {
    // V stored transposed per head: Vt[b][h][d][s]; per-lane contiguous v8h store
    const int b = row0 >> 11, s0 = row0 & 2047;
    v8h p0, p1;
#pragma unroll
    for (int r = 0; r < 8; ++r) { p0[r] = (_Float16)acc0[r]; p1[r] = (_Float16)acc1[r]; }
    const int n1 = n0 + lr, n2 = n1 + 512;
    *(v8h*)(Vt + ((size_t)(b * 16 + (n1 >> 6)) * 64 + (n1 & 63)) * 2048 + s0) = p0;
    *(v8h*)(Vt + ((size_t)(b * 16 + (n2 >> 6)) * 64 + (n2 & 63)) * 2048 + s0) = p1;
  }
}

// --------------------------------------------------------------------------
// Flash attention with causal mask + ALiBi.
// grid: (16 q-blocks of 128, 16 heads, 2 batches), block 256 (8 waves).
// Each wave: 16-query strip, streams keys in chunks of 32 with online softmax.
// (No cross-wave barriers: per-wave trip counts differ under the causal bound.)
// --------------------------------------------------------------------------
__global__ __launch_bounds__(256)
void attn_kernel(const _Float16* __restrict__ Qr,
                 const _Float16* __restrict__ Kr,
                 const _Float16* __restrict__ Vt,
                 _Float16* __restrict__ AO) {
  __shared__ __align__(64) _Float16 plds[8][512];  // per-wave 16x32 P tile
  const int lane = threadIdx.x & 31, wave = threadIdx.x >> 5;
  const int lr = lane & 15, lh = lane >> 4;
  const int b = blockIdx.z, h = blockIdx.y;
  const int qs = blockIdx.x * 128 + wave * 16;

  const _Float16* Qp = Qr + (size_t)(b * 2048) * 1024 + h * 64;
  const _Float16* Kp = Kr + (size_t)(b * 2048) * 1024 + h * 64;
  const _Float16* Vp = Vt + (size_t)((b * 16 + h) * 64) * 2048;

  // Q A-fragments for d=0..31 and d=32..63 (fixed for whole key loop)
  AFrag a0, a1;
  {
    const _Float16* q = Qp + (size_t)(qs + lr) * 1024 + lh * 16;
    a0.h.lo = *(const v8h*)(q);      a0.h.hi = *(const v8h*)(q + 8);
    a1.h.lo = *(const v8h*)(q + 32); a1.h.hi = *(const v8h*)(q + 40);
  }

  float mrow[8], lrow[8];
  v8f O0 = {}, O1 = {}, O2 = {}, O3 = {};
#pragma unroll
  for (int r = 0; r < 8; ++r) { mrow[r] = -3.0e38f; lrow[r] = 0.0f; }
  const float slope = exp2f(-0.5f * (float)(h + 1));   // ALiBi slope 2^(-8(h+1)/16)

  _Float16* myp = &plds[wave][0];

  for (int kb = 0; kb < qs + 16; kb += 32) {
    // ---- scores: 16x32 = two 16x16 WMMA accum tiles over K(d)=64 ----
    const int k0 = kb + lr, k1 = k0 + 16;
    const _Float16* kp0 = Kp + (size_t)k0 * 1024 + lh * 16;
    const _Float16* kp1 = Kp + (size_t)k1 * 1024 + lh * 16;
    v16h bk00 = *(const v16h*)(kp0);
    v16h bk01 = *(const v16h*)(kp0 + 32);
    v16h bk10 = *(const v16h*)(kp1);
    v16h bk11 = *(const v16h*)(kp1 + 32);
    v8f c0 = {}, c1 = {};
    c0 = WMMA_F16(a0.v, bk00, c0);
    c0 = WMMA_F16(a1.v, bk01, c0);
    c1 = WMMA_F16(a0.v, bk10, c1);
    c1 = WMMA_F16(a1.v, bk11, c1);

    // ---- scale + ALiBi + causal mask, chunk row-max ----
    float s0a[8], s1a[8], mx[8];
#pragma unroll
    for (int r = 0; r < 8; ++r) {
      const float qi = (float)(qs + r + 8 * lh);
      float v0 = c0[r] * 0.125f + slope * ((float)k0 - qi);
      float v1 = c1[r] * 0.125f + slope * ((float)k1 - qi);
      if ((float)k0 > qi) v0 = -1.0e9f;
      if ((float)k1 > qi) v1 = -1.0e9f;
      s0a[r] = v0; s1a[r] = v1;
      mx[r] = fmaxf(v0, v1);
    }
#pragma unroll
    for (int msk = 1; msk < 16; msk <<= 1) {
#pragma unroll
      for (int r = 0; r < 8; ++r) mx[r] = fmaxf(mx[r], __shfl_xor(mx[r], msk, 32));
    }

    // ---- online softmax update ----
    float alpha[8], rs[8];
#pragma unroll
    for (int r = 0; r < 8; ++r) {
      const float mn = fmaxf(mrow[r], mx[r]);
      alpha[r] = __expf(mrow[r] - mn);
      mrow[r] = mn;
    }
#pragma unroll
    for (int r = 0; r < 8; ++r) {
      const float p0 = __expf(s0a[r] - mrow[r]);
      const float p1 = __expf(s1a[r] - mrow[r]);
      s0a[r] = p0; s1a[r] = p1;
      rs[r] = p0 + p1;
    }
#pragma unroll
    for (int msk = 1; msk < 16; msk <<= 1) {
#pragma unroll
      for (int r = 0; r < 8; ++r) rs[r] += __shfl_xor(rs[r], msk, 32);
    }
#pragma unroll
    for (int r = 0; r < 8; ++r) {
      lrow[r] = lrow[r] * alpha[r] + rs[r];
      O0[r] *= alpha[r]; O1[r] *= alpha[r]; O2[r] *= alpha[r]; O3[r] *= alpha[r];
    }

    // ---- C-frag -> A-frag relayout of P through per-wave LDS ----
#pragma unroll
    for (int r = 0; r < 8; ++r) {
      const int rowi = r + 8 * lh;
      myp[rowi * 32 + lr]      = (_Float16)s0a[r];
      myp[rowi * 32 + lr + 16] = (_Float16)s1a[r];
    }
    asm volatile("s_wait_dscnt 0" ::: "memory");
    AFrag pf;
    {
      const _Float16* pp = myp + lr * 32 + lh * 16;
      pf.h.lo = *(const v8h*)pp;
      pf.h.hi = *(const v8h*)(pp + 8);
    }

    // ---- O += P @ V (V transposed -> contiguous B-frag loads) ----
    const _Float16* vb = Vp + (size_t)lr * 2048 + kb + lh * 16;
    v16h v0f = *(const v16h*)(vb);
    v16h v1f = *(const v16h*)(vb + (size_t)16 * 2048);
    v16h v2f = *(const v16h*)(vb + (size_t)32 * 2048);
    v16h v3f = *(const v16h*)(vb + (size_t)48 * 2048);
    O0 = WMMA_F16(pf.v, v0f, O0);
    O1 = WMMA_F16(pf.v, v1f, O1);
    O2 = WMMA_F16(pf.v, v2f, O2);
    O3 = WMMA_F16(pf.v, v3f, O3);
  }

  // ---- normalize and store per-head output ----
  float inv[8];
#pragma unroll
  for (int r = 0; r < 8; ++r) inv[r] = 1.0f / lrow[r];
  _Float16* out = AO + (size_t)(b * 2048 + qs) * 1024 + h * 64;
#pragma unroll
  for (int r = 0; r < 8; ++r) {
    const int ro = (r + 8 * lh) * 1024;
    out[ro + lr]      = (_Float16)(O0[r] * inv[r]);
    out[ro + 16 + lr] = (_Float16)(O1[r] * inv[r]);
    out[ro + 32 + lr] = (_Float16)(O2[r] * inv[r]);
    out[ro + 48 + lr] = (_Float16)(O3[r] * inv[r]);
  }
}

// --------------------------------------------------------------------------
// Output projection: out = AO @ out_w.T + out_b  (f32 result)
// grid: (32 row tiles of 128, 32 col tiles of 32), block 256 (8 waves).
// Weight tile staged via async-to-LDS, double buffered (shared by all waves).
// --------------------------------------------------------------------------
__global__ __launch_bounds__(256)
void oproj_kernel(const _Float16* __restrict__ AO,
                  const _Float16* __restrict__ owh,
                  const float* __restrict__ bias,
                  float* __restrict__ out) {
  __shared__ __align__(128) _Float16 bstage[2][32][32];
  const int tid  = threadIdx.x;
  const int lane = tid & 31;
  const int wave = tid >> 5;
  const int lr = lane & 15, lh = lane >> 4;
  const int rowbase = blockIdx.x * 128 + wave * 16;
  const int n0 = blockIdx.y * 32;

  const _Float16* aptr = AO + (size_t)(rowbase + lr) * 1024 + lh * 16;

  const bool cp  = tid < 128;
  const int crow = tid >> 2;
  const int cc   = (tid & 3) * 8;
  const _Float16* gsrc = owh + (size_t)(n0 + crow) * 1024 + cc;

  if (cp) async_g2l_b128(gsrc, &bstage[0][crow][cc]);

  v8f acc0 = {}, acc1 = {};
  for (int kk = 0; kk < 1024; kk += 32) {
    const int cur = (kk >> 5) & 1;
    if (cp) {
      if (kk + 32 < 1024) {
        async_g2l_b128(gsrc + kk + 32, &bstage[cur ^ 1][crow][cc]);
        asm volatile("s_wait_asynccnt 0x1" ::: "memory");
      } else {
        asm volatile("s_wait_asynccnt 0x0" ::: "memory");
      }
    }
    __syncthreads();

    __builtin_prefetch(aptr + kk + 256, 0, 3);
    AFrag a;
    a.h.lo = *(const v8h*)(aptr + kk);
    a.h.hi = *(const v8h*)(aptr + kk + 8);
    v16h b0 = *(const v16h*)&bstage[cur][lr][lh * 16];
    v16h b1 = *(const v16h*)&bstage[cur][16 + lr][lh * 16];
    acc0 = WMMA_F16(a.v, b0, acc0);
    acc1 = WMMA_F16(a.v, b1, acc1);
    __syncthreads();
  }

  const int col0 = n0 + lr, col1 = col0 + 16;
  const float bb0 = bias[col0], bb1 = bias[col1];
  const int row0 = rowbase + 8 * lh;
#pragma unroll
  for (int r = 0; r < 8; ++r) {
    out[(size_t)(row0 + r) * 1024 + col0] = acc0[r] + bb0;
    out[(size_t)(row0 + r) * 1024 + col1] = acc1[r] + bb1;
  }
}

// --------------------------------------------------------------------------
// Host-side launch
// --------------------------------------------------------------------------
extern "C" void kernel_launch(void* const* d_in, const int* in_sizes, int n_in,
                              void* d_out, int out_size, void* d_ws, size_t ws_size,
                              hipStream_t stream) {
  (void)in_sizes; (void)n_in; (void)out_size; (void)ws_size;
  const float* x   = (const float*)d_in[0];
  const float* wq  = (const float*)d_in[1];
  const float* wk  = (const float*)d_in[2];
  const float* wv  = (const float*)d_in[3];
  const float* ow  = (const float*)d_in[4];
  const float* ob  = (const float*)d_in[5];

  constexpr size_t MD = 1024 * 1024;       // 1M elements (one DxD matrix)
  _Float16* W   = (_Float16*)d_ws;
  _Float16* xh  = W;                       // 4M elems
  _Float16* wqh = xh  + 4 * MD;
  _Float16* wkh = wqh + MD;
  _Float16* wvh = wkh + MD;
  _Float16* owh = wvh + MD;
  _Float16* Qr  = owh + MD;                // 4M (roped Q, f16, [4096][1024])
  _Float16* Kr  = Qr  + 4 * MD;            // 4M (roped K)
  _Float16* Vt  = Kr  + 4 * MD;            // 4M (V transposed per head)
  _Float16* AO  = Vt  + 4 * MD;            // 4M (attention output, f16)

  cvt_f16_kernel<<<512, 256, 0, stream>>>(x,  xh,  4096 * 1024);
  cvt_f16_kernel<<<256, 256, 0, stream>>>(wq, wqh, 1024 * 1024);
  cvt_f16_kernel<<<256, 256, 0, stream>>>(wk, wkh, 1024 * 1024);
  cvt_f16_kernel<<<256, 256, 0, stream>>>(wv, wvh, 1024 * 1024);
  cvt_f16_kernel<<<256, 256, 0, stream>>>(ow, owh, 1024 * 1024);

  qkv_kernel<<<dim3(32, 32, 3), 256, 0, stream>>>(xh, wqh, wkh, wvh, Qr, Kr, Vt);
  attn_kernel<<<dim3(16, 16, 2), 256, 0, stream>>>(Qr, Kr, Vt, AO);
  oproj_kernel<<<dim3(32, 32), 256, 0, stream>>>(AO, owh, ob, (float*)d_out);
}
